// EnhancedBilinearInteraction_79345225826311
// MI455X (gfx1250) — matrix operand order repaired and checked by reading, compiled, not run on hardware
//
#include <hip/hip_runtime.h>
#include <hip/hip_bf16.h>
#include <math.h>

// ---------------------------------------------------------------------------
// Problem constants (from reference): B=32, L=2048, C_IN=256, HID=512
// out[b,h] = sum_l tanh( (W2 @ (W1 @ x[b,l,:] + b1) + b2)[h] )
//                * tanh( (W2 @ (W1 @ y[b,l,:] + b1) + b2)[h] )
// x and y share all weights -> process both in one flow, sharing B fragments.
// ---------------------------------------------------------------------------
#define BB   32
#define LL   2048
#define CIN  256
#define HID  512
#define LT   64                    // L-tile rows per workgroup

// LDS layout (byte strides padded +16B so 16-row B128 column reads hit
// distinct bank groups: stride/4 mod 64 == 4 banks shift per row)
#define IN_S    528                // 256 bf16 + 8 pad
#define H1_S    1040               // 512 bf16 + 8 pad
#define OFF_INX 0
#define OFF_INY (LT * IN_S)                  // 33792
#define OFF_H1X (2 * LT * IN_S)              // 67584
#define OFF_H1Y (2 * LT * IN_S + LT * H1_S)  // 134144
#define LDS_BYTES (2 * LT * IN_S + 2 * LT * H1_S)  // 200704 -> 1 WG / WGP

typedef __bf16 bf16;
typedef __attribute__((ext_vector_type(16))) __bf16 v16bf;
typedef __attribute__((ext_vector_type(8)))  __bf16 v8bf;
typedef __attribute__((ext_vector_type(8)))  float  v8f;

// CDNA5 hardware TANH transcendental (V_TANH_F32) co-executes with WMMA/XDL.
#if __has_builtin(__builtin_amdgcn_tanhf)
__device__ __forceinline__ float fast_tanh(float v) { return __builtin_amdgcn_tanhf(v); }
#else
__device__ __forceinline__ float fast_tanh(float v) { return tanhf(v); }
#endif

// Native bf16 converts (gfx1250: v_cvt_pk_bf16_f32, RNE).
__device__ __forceinline__ unsigned pack2_bf16(float a, float b) {
  union { __bf16 h[2]; unsigned u; } c;
  c.h[0] = (__bf16)a;
  c.h[1] = (__bf16)b;
  return c.u;
}

// ---------------------------------------------------------------------------
// Fused kernel: grid = B * (L/LT) = 1024 blocks, 256 threads (8 waves).
// Wave w owns output columns h in [64w, 64w+64) as 4 N-tiles; all 4 M-tiles.
// X and Y accumulator chains share one B-fragment set per k-step, so all
// outer loops can stay rolled: no private arrays, ~200 VGPRs, no spills.
// ---------------------------------------------------------------------------
__global__ __launch_bounds__(256) void fused_proj_tanh_dot(
    const float* __restrict__ x, const float* __restrict__ y,
    const bf16*  __restrict__ W1b, const bf16* __restrict__ W2b,
    const float* __restrict__ b1,  const float* __restrict__ b2,
    float* __restrict__ out)
{
  extern __shared__ char smem[];
  const int tid   = threadIdx.x;
  const int wave  = tid >> 5;
  const int lane  = tid & 31;
  const int bidx  = blockIdx.x >> 5;        // batch
  const int lt    = blockIdx.x & 31;        // L-tile index
  const int arow  = lane & 15;              // A fragment: M row within tile
  const int kst   = (lane >> 4) << 3;       // A fragment: k start (0 or 8)
  const int koffB = (lane >> 4) << 3;       // B fragment: k start (0 or 8)

  const float* xsrc = x + ((size_t)bidx * LL + (size_t)lt * LT) * CIN;
  const float* ysrc = y + ((size_t)bidx * LL + (size_t)lt * LT) * CIN;

  // ---- stage x AND y tiles [64 x 256] f32 -> bf16 in LDS (coalesced) ----
  {
    const float4* sx = (const float4*)xsrc;
    const float4* sy = (const float4*)ysrc;
    #pragma unroll
    for (int i = 0; i < 16; ++i) {
      int q   = i * 256 + tid;              // 4096 float4 per tensor
      int row = q >> 6;                     // 64 float4 per row
      int c4  = q & 63;
      float4 vx = sx[q];
      float4 vy = sy[q];
      uint2 px, py;
      px.x = pack2_bf16(vx.x, vx.y);  px.y = pack2_bf16(vx.z, vx.w);
      py.x = pack2_bf16(vy.x, vy.y);  py.y = pack2_bf16(vy.z, vy.w);
      *(uint2*)(smem + OFF_INX + row * IN_S + c4 * 8) = px;
      *(uint2*)(smem + OFF_INY + row * IN_S + c4 * 8) = py;
    }
  }
  __syncthreads();

  // ---- GEMM1 (x & y together): h1 = in * W1^T + b1  (bf16 -> LDS) ----
  #pragma unroll 1
  for (int j = 0; j < 4; ++j) {
    const int   hcol = (wave * 4 + j) * 16 + arow;     // per-lane output column
    const float bias = b1[hcol];
    const bf16* brow = W1b + (size_t)hcol * CIN + koffB;
    #pragma unroll 1
    for (int m = 0; m < 4; ++m) {
      const int   rowoff = (m * 16 + arow) * IN_S + kst * 2;
      const char* ax = smem + OFF_INX + rowoff;
      const char* ay = smem + OFF_INY + rowoff;
      v8f accX = {}, accY = {};
      #pragma unroll
      for (int k0 = 0; k0 < CIN; k0 += 32) {
        v8bf blo = *(const v8bf*)(brow + k0);          // shared B fragment
        v8bf bhi = *(const v8bf*)(brow + k0 + 16);
        v8bf xlo = *(const v8bf*)(ax + k0 * 2);
        v8bf xhi = *(const v8bf*)(ax + k0 * 2 + 32);
        v8bf ylo = *(const v8bf*)(ay + k0 * 2);
        v8bf yhi = *(const v8bf*)(ay + k0 * 2 + 32);
        v16bf bm, aX, aY;
        #pragma unroll
        for (int e = 0; e < 8; ++e) {
          bm[e] = blo[e]; bm[e + 8] = bhi[e];
          aX[e] = xlo[e]; aX[e + 8] = xhi[e];
          aY[e] = ylo[e]; aY[e + 8] = yhi[e];
        }
        accX = __builtin_amdgcn_wmma_f32_16x16x32_bf16(
                   false, aX, false, bm, (short)0, accX, false, false);
        accY = __builtin_amdgcn_wmma_f32_16x16x32_bf16(
                   false, aY, false, bm, (short)0, accY, false, false);
      }
      const int r0 = m * 16 + ((lane >> 4) << 3);      // D rows: r0..r0+7
      #pragma unroll
      for (int r = 0; r < 8; ++r) {
        const size_t off = (size_t)(r0 + r) * H1_S + hcol * 2;
        *(__bf16*)(smem + OFF_H1X + off) = (__bf16)(accX[r] + bias);
        *(__bf16*)(smem + OFF_H1Y + off) = (__bf16)(accY[r] + bias);
      }
    }
  }
  __syncthreads();

  // ---- GEMM2 (x & y together) + tanh product reduction ----
  #pragma unroll 1
  for (int j = 0; j < 4; ++j) {
    const int   hcol = (wave * 4 + j) * 16 + arow;
    const float bias = b2[hcol];
    const bf16* brow = W2b + (size_t)hcol * HID + koffB;
    float partial = 0.f;
    #pragma unroll 1
    for (int m = 0; m < 4; ++m) {
      const int   rowoff = (m * 16 + arow) * H1_S + kst * 2;
      const char* ax = smem + OFF_H1X + rowoff;
      const char* ay = smem + OFF_H1Y + rowoff;
      v8f accX = {}, accY = {};
      #pragma unroll
      for (int k0 = 0; k0 < HID; k0 += 32) {
        v8bf blo = *(const v8bf*)(brow + k0);          // shared B fragment
        v8bf bhi = *(const v8bf*)(brow + k0 + 16);
        v8bf xlo = *(const v8bf*)(ax + k0 * 2);
        v8bf xhi = *(const v8bf*)(ax + k0 * 2 + 32);
        v8bf ylo = *(const v8bf*)(ay + k0 * 2);
        v8bf yhi = *(const v8bf*)(ay + k0 * 2 + 32);
        v16bf bm, aX, aY;
        #pragma unroll
        for (int e = 0; e < 8; ++e) {
          bm[e] = blo[e]; bm[e + 8] = bhi[e];
          aX[e] = xlo[e]; aX[e + 8] = xhi[e];
          aY[e] = ylo[e]; aY[e + 8] = yhi[e];
        }
        accX = __builtin_amdgcn_wmma_f32_16x16x32_bf16(
                   false, aX, false, bm, (short)0, accX, false, false);
        accY = __builtin_amdgcn_wmma_f32_16x16x32_bf16(
                   false, aY, false, bm, (short)0, accY, false, false);
      }
      #pragma unroll
      for (int r = 0; r < 8; ++r)
        partial += fast_tanh(accX[r] + bias) * fast_tanh(accY[r] + bias);
    }
    atomicAdd(out + (size_t)bidx * HID + hcol, partial);
  }
}

// ---------------------------------------------------------------------------
// Prep kernels
// ---------------------------------------------------------------------------
__global__ void cvt_bf16_kernel(const float* __restrict__ s,
                                bf16* __restrict__ d, int n) {
  int i = blockIdx.x * 256 + threadIdx.x;
  if (i < n) d[i] = (__bf16)s[i];
}

__global__ void zero_kernel(float* __restrict__ p, int n) {
  int i = blockIdx.x * 256 + threadIdx.x;
  if (i < n) p[i] = 0.f;
}

// ---------------------------------------------------------------------------
extern "C" void kernel_launch(void* const* d_in, const int* in_sizes, int n_in,
                              void* d_out, int out_size, void* d_ws, size_t ws_size,
                              hipStream_t stream) {
  const float* x  = (const float*)d_in[0];
  const float* y  = (const float*)d_in[1];
  const float* W1 = (const float*)d_in[2];
  const float* b1 = (const float*)d_in[3];
  const float* W2 = (const float*)d_in[4];
  const float* b2 = (const float*)d_in[5];
  float* out = (float*)d_out;

  // workspace: bf16 copies of W1 (256 KB) and W2 (512 KB)
  bf16* W1b = (bf16*)d_ws;
  bf16* W2b = W1b + (size_t)HID * CIN;

  cvt_bf16_kernel<<<(HID * CIN + 255) / 256, 256, 0, stream>>>(W1, W1b, HID * CIN);
  cvt_bf16_kernel<<<(HID * HID + 255) / 256, 256, 0, stream>>>(W2, W2b, HID * HID);
  zero_kernel<<<(BB * HID + 255) / 256, 256, 0, stream>>>(out, BB * HID);

  fused_proj_tanh_dot<<<dim3(BB * (LL / LT)), dim3(256), LDS_BYTES, stream>>>(
      x, y, (const bf16*)W1b, (const bf16*)W2b, b1, b2, out);
}
